// SelfAttention_33277406609963
// MI455X (gfx1250) — matrix operand order repaired and checked by reading, not compiled, not run on hardware
//
#include <hip/hip_runtime.h>
#include <hip/hip_fp16.h>

typedef _Float16 f16t;
typedef __attribute__((ext_vector_type(16))) _Float16 v16h;
typedef __attribute__((ext_vector_type(8)))  float    v8f;

union Frag16 { v16h v; _Float16 h[16]; unsigned u[8]; uint4 q[2]; };
union FragC  { v8f  v; float f[8]; };

#define B_     4
#define C_     128
#define C8_    16
#define N_     4096
#define JCHUNK 64
#define NCHUNK (N_ / JCHUNK)

// ---------------------------------------------------------------------------
// DPP16 row reductions (16-lane halves): quad_perm xor1 (0xB1), xor2 (0x4E),
// row_half_mirror (0x141), row_mirror (0x140). Pure VALU, no LDS.
// ---------------------------------------------------------------------------
template <int CTRL>
__device__ __forceinline__ float dppmov(float x) {
    int xi = __builtin_bit_cast(int, x);
    int r  = __builtin_amdgcn_update_dpp(xi, xi, CTRL, 0xF, 0xF, true);
    return __builtin_bit_cast(float, r);
}
__device__ __forceinline__ float red16_max(float x) {
    x = fmaxf(x, dppmov<0xB1>(x));
    x = fmaxf(x, dppmov<0x4E>(x));
    x = fmaxf(x, dppmov<0x141>(x));
    x = fmaxf(x, dppmov<0x140>(x));
    return x;
}
__device__ __forceinline__ float red16_sum(float x) {
    x += dppmov<0xB1>(x);
    x += dppmov<0x4E>(x);
    x += dppmov<0x141>(x);
    x += dppmov<0x140>(x);
    return x;
}

// ---------------------------------------------------------------------------
// gfx1250 async global->LDS staging (ASYNCcnt-tracked).
// LDS address = low 32 bits of generic pointer to __shared__.
// ---------------------------------------------------------------------------
__device__ __forceinline__ unsigned lds_off(const void* p) {
    return (unsigned)(uintptr_t)p;
}
__device__ __forceinline__ void async_b128(unsigned lds, unsigned voff,
                                           unsigned long long base) {
    asm volatile("global_load_async_to_lds_b128 %0, %1, %2"
                 :: "v"(lds), "v"(voff), "s"(base) : "memory");
}
__device__ __forceinline__ void async_b64(unsigned lds, unsigned voff,
                                          unsigned long long base) {
    asm volatile("global_load_async_to_lds_b64 %0, %1, %2"
                 :: "v"(lds), "v"(voff), "s"(base) : "memory");
}

// Stage one 64-key chunk: K (2KB) + V (16KB) -> 5 async ops per thread/wave.
__device__ __forceinline__ void stage_chunk(int tid,
                                            unsigned long long kbase,
                                            unsigned long long vbase,
                                            unsigned sK_lds, unsigned sV_lds,
                                            int j0) {
    // K: [b][n][c8] rows are contiguous -> straight 2KB copy, 8B/thread
    async_b64(sK_lds + tid * 8, (unsigned)(j0 * 32 + tid * 8), kbase);
    // V: [b][c][n]; per c copy 64 halves (128B); 64B per thread
    int c   = tid >> 1;
    int seg = tid & 1;
    unsigned gof = (unsigned)((c * N_ + j0) * 2 + seg * 64);
    unsigned lof = sV_lds + (unsigned)(c * 128 + seg * 64);
    async_b128(lof +  0, gof +  0, vbase);
    async_b128(lof + 16, gof + 16, vbase);
    async_b128(lof + 32, gof + 32, vbase);
    async_b128(lof + 48, gof + 48, vbase);
}

// ---------------------------------------------------------------------------
// Pass 1a: q/k projections -> f16 transposed [b][n][c8]
// ---------------------------------------------------------------------------
__global__ void proj_qk_kernel(const float* __restrict__ x,
                               const float* __restrict__ Wq, const float* __restrict__ bq,
                               const float* __restrict__ Wk, const float* __restrict__ bk,
                               f16t* __restrict__ qT, f16t* __restrict__ kT) {
    int id = blockIdx.x * blockDim.x + threadIdx.x;
    int i  = id & (N_ - 1);
    int t  = id >> 12;
    int c8 = t & 15;  t >>= 4;
    int b  = t & 3;   t >>= 2;
    int which = t;
    const float* W    = which ? Wk : Wq;
    const float* bias = which ? bk : bq;
    float acc = bias[c8];
    const float* xp = x + (size_t)b * C_ * N_ + i;
    const float* wp = W + c8 * C_;
#pragma unroll 8
    for (int c = 0; c < C_; ++c) acc += wp[c] * xp[(size_t)c * N_];
    f16t* dst = which ? kT : qT;
    dst[((size_t)b * N_ + i) * C8_ + c8] = (f16t)acc;
}

// ---------------------------------------------------------------------------
// Pass 1b: v projection -> f16 [b][c][n]
// ---------------------------------------------------------------------------
__global__ void proj_v_kernel(const float* __restrict__ x,
                              const float* __restrict__ Wv, const float* __restrict__ bv,
                              f16t* __restrict__ vO) {
    int id = blockIdx.x * blockDim.x + threadIdx.x;
    int i  = id & (N_ - 1);
    int t  = id >> 12;
    int c  = t & 127;
    int b  = t >> 7;
    float acc = bv[c];
    const float* xp = x + (size_t)b * C_ * N_ + i;
    const float* wp = Wv + c * C_;
#pragma unroll 8
    for (int cc = 0; cc < C_; ++cc) acc += wp[cc] * xp[(size_t)cc * N_];
    vO[(size_t)id] = (f16t)acc;
}

// ---------------------------------------------------------------------------
// Pass 2: flash attention, 64-key chunks, async double-buffered K/V staging.
// Block = 256 threads = 8 waves, one 16-query tile per wave.
// ---------------------------------------------------------------------------
__global__ __launch_bounds__(256)
void attn_kernel(const f16t* __restrict__ qT, const f16t* __restrict__ kT,
                 const f16t* __restrict__ vW, const float* __restrict__ x,
                 const float* __restrict__ gamma_p, float* __restrict__ out) {
    __shared__ __align__(16) _Float16 sK[2][JCHUNK * C8_];   //  4 KB [buf][j][c8]
    __shared__ __align__(16) _Float16 sV[2][C_ * JCHUNK];    // 32 KB [buf][c][j]
    __shared__ __align__(16) _Float16 sP[8][16 * JCHUNK];    // 16 KB [wave][m][j]

    const int tid  = threadIdx.x;
    const int b    = blockIdx.x >> 5;
    const int qg   = blockIdx.x & 31;
    const int w    = tid >> 5;
    const int lane = tid & 31;
    const int g    = lane >> 4;
    const int ln   = lane & 15;
    const int i0   = (qg * 8 + w) * 16;
    const float gamma = gamma_p[0];

    const unsigned long long kbase =
        (unsigned long long)(uintptr_t)(kT + (size_t)b * N_ * C8_);
    const unsigned long long vbase =
        (unsigned long long)(uintptr_t)(vW + (size_t)b * C_ * N_);
    const unsigned sK0 = lds_off(&sK[0][0]), sK1 = lds_off(&sK[1][0]);
    const unsigned sV0 = lds_off(&sV[0][0]), sV1 = lds_off(&sV[1][0]);

    // ---- Q A-fragment (K = channel 0..15, zero-padded to 32) ----
    Frag16 qa;
    {
        const uint4* qrow =
            (const uint4*)(qT + ((size_t)b * N_ + i0 + ln) * C8_);
        qa.q[0] = qrow[g];          // u32 idx 4g..4g+3  (c pairs 0..7 / 8..15)
        qa.q[1] = make_uint4(0u, 0u, 0u, 0u);
    }

    FragC O[8];
#pragma unroll
    for (int cg = 0; cg < 8; ++cg) O[cg].v = (v8f){};
    float mrow[8], lrow[8];
#pragma unroll
    for (int r = 0; r < 8; ++r) { mrow[r] = -1.0e30f; lrow[r] = 0.0f; }

    stage_chunk(tid, kbase, vbase, sK0, sV0, 0);   // prime buffer 0

    for (int ch = 0; ch < NCHUNK; ++ch) {
        const int cur = ch & 1;
        if (ch + 1 < NCHUNK) {
            stage_chunk(tid, kbase, vbase, cur ? sK0 : sK1, cur ? sV0 : sV1,
                        (ch + 1) * JCHUNK);
            asm volatile("s_wait_asynccnt 0x5" ::: "memory"); // chunk ch retired
        } else {
            asm volatile("s_wait_asynccnt 0x0" ::: "memory");
        }
        __syncthreads();
        const _Float16* Kc = sK[cur];
        const _Float16* Vc = sV[cur];

        // ---- S = Q^T K, four 16-key subtiles ----
        FragC S[4];
#pragma unroll
        for (int s = 0; s < 4; ++s) {
            Frag16 kb;
            kb.q[0] = make_uint4(0u, 0u, 0u, 0u);
            kb.q[1] = make_uint4(0u, 0u, 0u, 0u);
            {   // half g carries channels 8g..8g+7; K=16..31 -> zero pad
                const uint4* row = (const uint4*)(Kc + (16 * s + ln) * C8_);
                kb.q[0] = row[g];
            }
            S[s].v = __builtin_amdgcn_wmma_f32_16x16x32_f16(
                false, qa.v, false, kb.v, (short)0, (v8f){}, false, false);
        }

        // ---- online softmax (DPP row reductions, no LDS) ----
        float rmax[8], alpha[8], rsum[8];
#pragma unroll
        for (int r = 0; r < 8; ++r) {
            float mx = fmaxf(fmaxf(S[0].f[r], S[1].f[r]),
                             fmaxf(S[2].f[r], S[3].f[r]));
            rmax[r] = red16_max(mx);
        }
#pragma unroll
        for (int r = 0; r < 8; ++r) {
            float mn = fmaxf(mrow[r], rmax[r]);
            alpha[r] = __expf(mrow[r] - mn);
            mrow[r]  = mn;
            rsum[r]  = 0.0f;
        }
#pragma unroll
        for (int s = 0; s < 4; ++s)
#pragma unroll
            for (int r = 0; r < 8; ++r) {
                float p = __expf(S[s].f[r] - mrow[r]);
                rsum[r] += p;
                sP[w][(r + 8 * g) * JCHUNK + 16 * s + ln] = (f16t)p;
            }
#pragma unroll
        for (int r = 0; r < 8; ++r) {
            lrow[r] = lrow[r] * alpha[r] + red16_sum(rsum[r]);
        }
#pragma unroll
        for (int cg = 0; cg < 8; ++cg)
#pragma unroll
            for (int r = 0; r < 8; ++r) O[cg].f[r] *= alpha[r];

        // ---- P A-fragments (same-wave LDS round trip; DS is in-order) ----
        Frag16 pa[2];
        {
            const uint4* prow = (const uint4*)(sP[w] + ln * JCHUNK);
#pragma unroll
            for (int jh = 0; jh < 2; ++jh) {
                pa[jh].q[0] = prow[jh * 4 + g];
                pa[jh].q[1] = prow[jh * 4 + 2 + g];
            }
        }

        // ---- O += P * V^T : 8 channel groups x 2 K-halves ----
#pragma unroll
        for (int cg = 0; cg < 8; ++cg) {
            const uint4* vrow = (const uint4*)(Vc + (cg * 16 + ln) * JCHUNK);
#pragma unroll
            for (int jh = 0; jh < 2; ++jh) {
                Frag16 vb;
                vb.q[0] = vrow[jh * 4 + g];
                vb.q[1] = vrow[jh * 4 + 2 + g];
                O[cg].v = __builtin_amdgcn_wmma_f32_16x16x32_f16(
                    false, pa[jh].v, false, vb.v, (short)0, O[cg].v,
                    false, false);
            }
        }
        __syncthreads();   // all waves done with buf[cur] before restage
    }

    // ---- epilogue: out = gamma * O/l + x ----
#pragma unroll
    for (int cg = 0; cg < 8; ++cg) {
        int c = cg * 16 + ln;
#pragma unroll
        for (int r = 0; r < 8; ++r) {
            int i = i0 + r + 8 * g;
            size_t idx = ((size_t)b * C_ + c) * N_ + i;
            out[idx] = gamma * (O[cg].f[r] / lrow[r]) + x[idx];
        }
    }
}

// ---------------------------------------------------------------------------
extern "C" void kernel_launch(void* const* d_in, const int* in_sizes, int n_in,
                              void* d_out, int out_size, void* d_ws, size_t ws_size,
                              hipStream_t stream) {
    const float* x     = (const float*)d_in[0];
    const float* Wq    = (const float*)d_in[1];
    const float* bq    = (const float*)d_in[2];
    const float* Wk    = (const float*)d_in[3];
    const float* bk    = (const float*)d_in[4];
    const float* Wv    = (const float*)d_in[5];
    const float* bv    = (const float*)d_in[6];
    const float* gamma = (const float*)d_in[7];

    f16t* qT = (f16t*)d_ws;                         // [B][N][C8]  512 KB
    f16t* kT = qT + (size_t)B_ * N_ * C8_;          // [B][N][C8]  512 KB
    f16t* vW = kT + (size_t)B_ * N_ * C8_;          // [B][C][N]   4 MB

    proj_qk_kernel<<<(2 * B_ * C8_ * N_) / 256, 256, 0, stream>>>(
        x, Wq, bq, Wk, bk, qT, kT);
    proj_v_kernel<<<(B_ * C_ * N_) / 256, 256, 0, stream>>>(x, Wv, bv, vW);
    attn_kernel<<<B_ * (N_ / 16 / 8), 256, 0, stream>>>(
        qT, kT, vW, x, gamma, (float*)d_out);
}
